// GAT2_39316130627596
// MI455X (gfx1250) — matrix operand order repaired and checked by reading, not compiled, not run on hardware
//
#include <hip/hip_runtime.h>
#include <hip/hip_bf16.h>

typedef __attribute__((ext_vector_type(16))) _Float16 v16h;
typedef __attribute__((ext_vector_type(8)))  float    v8f;

#define NNODES 100000
#define NEDGES 1600000
#define DIM    64
#define NTILES (NNODES / 16)      // 6250 row tiles of 16
#define NEG_SLOPE 0.2f

// ---------- float <-> order-preserving uint (for atomic max on floats) ----------
static __device__ __forceinline__ unsigned f32_sortable(float f) {
    unsigned u = __float_as_uint(f);
    return (u & 0x80000000u) ? ~u : (u | 0x80000000u);
}
static __device__ __forceinline__ float sortable_f32(unsigned u) {
    return (u & 0x80000000u) ? __uint_as_float(u ^ 0x80000000u) : __uint_as_float(~u);
}

// ---------- prepack 9 weight matrices (3 layers x {Wl,Wr,Ws}) into f16 B-fragments ----
// layout: packed[layer][which(3)][nt(4)][ks(2)][lane(32)][16 halves]
// B (32x16, f16): lanes 0-15 hold K=0-15, lanes 16-31 hold K=16-31; VGPR v -> K=2v,2v+1
__global__ void __launch_bounds__(256) gat_prepack(
    const float* __restrict__ Wl, const float* __restrict__ Wr,
    const float* __restrict__ Ws, _Float16* __restrict__ packed)
{
    int idx = blockIdx.x * blockDim.x + threadIdx.x;
    if (idx >= 9 * 4 * 2 * 32 * 16) return;
    int h    = idx & 15;
    int lane = (idx >> 4) & 31;
    int ks   = (idx >> 9) & 1;
    int nt   = (idx >> 10) & 3;
    int mat  = idx >> 12;                // 0..8
    int layer = mat / 3, which = mat % 3;
    const float* W = (which == 0) ? Wl : (which == 1) ? Wr : Ws;
    W += layer * DIM * DIM;
    int nloc = lane & 15, hi = lane >> 4;
    int v = h >> 1, p = h & 1;
    int k = ks * 32 + hi * 16 + 2 * v + p;
    int n = nt * 16 + nloc;
    packed[idx] = (_Float16)W[k * DIM + n];
}

// ---------- fused GEMM: xl = h@Wl+bl, xr = h@Wr+br, xs = h@Ws+bs (one wave = 16 rows) ----
__global__ void __launch_bounds__(256) gat_gemm3_wmma(
    const float* __restrict__ hin, const _Float16* __restrict__ packed,
    const float* __restrict__ bl, const float* __restrict__ br, const float* __restrict__ bs,
    float* __restrict__ xl, float* __restrict__ xr, float* __restrict__ xs)
{
    int wave = (int)((blockIdx.x * blockDim.x + threadIdx.x) >> 5);
    if (wave >= NTILES) return;
    int lane = threadIdx.x & 31;
    int m  = lane & 15;
    int hi = lane >> 4;
    int row0 = wave * 16;

    // A fragments (16x32 f16): VGPR v -> K = hi*16 + 2v + p
    v16h a0, a1;
    const float* rowp = hin + (size_t)(row0 + m) * DIM;
#pragma unroll
    for (int v = 0; v < 8; ++v) {
        int kb = hi * 16 + 2 * v;
        float2 f0 = *(const float2*)(rowp + kb);
        float2 f1 = *(const float2*)(rowp + 32 + kb);
        a0[2 * v]     = (_Float16)f0.x;
        a0[2 * v + 1] = (_Float16)f0.y;
        a1[2 * v]     = (_Float16)f1.x;
        a1[2 * v + 1] = (_Float16)f1.y;
    }

    float*       outs[3]   = {xl, xr, xs};
    const float* biases[3] = {bl, br, bs};
#pragma unroll
    for (int mat = 0; mat < 3; ++mat) {
        float* outp = outs[mat];
        const float* bias = biases[mat];
#pragma unroll
        for (int nt = 0; nt < 4; ++nt) {
            const _Float16* bp = packed + ((size_t)((mat * 4 + nt) * 2 + 0) * 32 + lane) * 16;
            v16h b0 = *(const v16h*)bp;
            v16h b1 = *(const v16h*)(bp + 32 * 16);
            v8f c = {};
            c = __builtin_amdgcn_wmma_f32_16x16x32_f16(false, a0, false, b0, (short)0, c, false, false);
            c = __builtin_amdgcn_wmma_f32_16x16x32_f16(false, a1, false, b1, (short)0, c, false, false);
            int n = nt * 16 + m;
            float bv = bias[n];
#pragma unroll
            for (int v = 0; v < 8; ++v)
                outp[(size_t)(row0 + hi * 8 + v) * DIM + n] = c[v] + bv;
        }
    }
}

// ---------- zero per-layer accumulators ----------
__global__ void __launch_bounds__(256) gat_init(
    float* __restrict__ agg, float* __restrict__ denom, unsigned* __restrict__ mmax)
{
    int i = blockIdx.x * blockDim.x + threadIdx.x;
    if (i < NNODES * DIM) agg[i] = 0.f;
    if (i < NNODES) { denom[i] = 0.f; mmax[i] = 0u; }   // 0u = sortable(-inf-ish minimum)
}

// ---------- per-edge attention logit + running per-dest max ----------
__global__ void __launch_bounds__(256) gat_edge_attn(
    const long long* __restrict__ ei, const float* __restrict__ xl,
    const float* __restrict__ xr, const float* __restrict__ att,
    float* __restrict__ ebuf, unsigned* __restrict__ mmax)
{
    int e = blockIdx.x * blockDim.x + threadIdx.x;
    if (e >= NEDGES) return;
    int s = (int)ei[e];
    int d = (int)ei[NEDGES + e];
    const float4* l = (const float4*)(xl + (size_t)s * DIM);
    const float4* r = (const float4*)(xr + (size_t)d * DIM);
    const float4* a = (const float4*)att;
    float acc = 0.f;
#pragma unroll
    for (int i = 0; i < 16; ++i) {
        float4 lv = l[i], rv = r[i], av = a[i];
        float vx = lv.x + rv.x, vy = lv.y + rv.y, vz = lv.z + rv.z, vw = lv.w + rv.w;
        vx = vx > 0.f ? vx : NEG_SLOPE * vx;
        vy = vy > 0.f ? vy : NEG_SLOPE * vy;
        vz = vz > 0.f ? vz : NEG_SLOPE * vz;
        vw = vw > 0.f ? vw : NEG_SLOPE * vw;
        acc += av.x * vx + av.y * vy + av.z * vz + av.w * vw;
    }
    ebuf[e] = acc;
    atomicMax(mmax + d, f32_sortable(acc));
}

// ---------- per-edge exp(e - max) + per-dest denominator ----------
__global__ void __launch_bounds__(256) gat_edge_exp(
    const long long* __restrict__ ei, const unsigned* __restrict__ mmax,
    float* __restrict__ ebuf, float* __restrict__ denom)
{
    int e = blockIdx.x * blockDim.x + threadIdx.x;
    if (e >= NEDGES) return;
    int d = (int)ei[NEDGES + e];
    float a = expf(ebuf[e] - sortable_f32(mmax[d]));
    ebuf[e] = a;
    atomicAdd(denom + d, a);
}

// ---------- wave-per-edge weighted scatter: agg[dst] += alpha * xl[src] ----------
__global__ void __launch_bounds__(256) gat_edge_agg(
    const long long* __restrict__ ei, const float* __restrict__ ebuf,
    const float* __restrict__ denom, const float* __restrict__ xl,
    float* __restrict__ agg)
{
    int e = (int)((blockIdx.x * blockDim.x + threadIdx.x) >> 5);
    if (e >= NEDGES) return;
    int lane = threadIdx.x & 31;
    int s = (int)ei[e];
    int d = (int)ei[NEDGES + e];
    float alpha = ebuf[e] / denom[d];
    float2 v = ((const float2*)(xl + (size_t)s * DIM))[lane];
    atomicAdd(agg + (size_t)d * DIM + 2 * lane,     alpha * v.x);
    atomicAdd(agg + (size_t)d * DIM + 2 * lane + 1, alpha * v.y);
}

// ---------- node post: h = [relu](agg + bg + xs) ----------
__global__ void __launch_bounds__(256) gat_node_post(
    const float* __restrict__ agg, const float* __restrict__ xs,
    const float* __restrict__ bg, float* __restrict__ hout, int do_relu)
{
    int i = blockIdx.x * blockDim.x + threadIdx.x;   // over N*16 float4s
    if (i >= NNODES * 16) return;
    int c4 = i & 15;
    float4 v  = ((const float4*)agg)[i];
    float4 sv = ((const float4*)xs)[i];
    float4 b  = ((const float4*)bg)[c4];
    float ox = v.x + sv.x + b.x, oy = v.y + sv.y + b.y;
    float oz = v.z + sv.z + b.z, ow = v.w + sv.w + b.w;
    if (do_relu) {
        ox = ox > 0.f ? ox : 0.f; oy = oy > 0.f ? oy : 0.f;
        oz = oz > 0.f ? oz : 0.f; ow = ow > 0.f ? ow : 0.f;
    }
    float4 o; o.x = ox; o.y = oy; o.z = oz; o.w = ow;
    ((float4*)hout)[i] = o;
}

// ---------- final projection: out = h @ Wout + bout ----------
__global__ void __launch_bounds__(256) gat_out_proj(
    const float* __restrict__ h, const float* __restrict__ Wout,
    const float* __restrict__ bout, float* __restrict__ out)
{
    int n = blockIdx.x * blockDim.x + threadIdx.x;
    if (n >= NNODES) return;
    const float* row = h + (size_t)n * DIM;
    float a0 = bout[0], a1 = bout[1];
#pragma unroll
    for (int k = 0; k < DIM; ++k) {
        float x = row[k];
        a0 += x * Wout[2 * k];
        a1 += x * Wout[2 * k + 1];
    }
    out[2 * n]     = a0;
    out[2 * n + 1] = a1;
}

extern "C" void kernel_launch(void* const* d_in, const int* in_sizes, int n_in,
                              void* d_out, int out_size, void* d_ws, size_t ws_size,
                              hipStream_t stream) {
    (void)in_sizes; (void)n_in; (void)out_size; (void)ws_size;
    const float*     x    = (const float*)d_in[0];
    const long long* ei   = (const long long*)d_in[1];
    const float*     Wl   = (const float*)d_in[2];
    const float*     bl   = (const float*)d_in[3];
    const float*     Wr   = (const float*)d_in[4];
    const float*     br   = (const float*)d_in[5];
    const float*     att  = (const float*)d_in[6];
    const float*     bg   = (const float*)d_in[7];
    const float*     Ws   = (const float*)d_in[8];
    const float*     bs   = (const float*)d_in[9];
    const float*     Wout = (const float*)d_in[10];
    const float*     bout = (const float*)d_in[11];
    float*           out  = (float*)d_out;

    const size_t NB = (size_t)NNODES * DIM * sizeof(float);   // 25.6 MB
    char* ws = (char*)d_ws;
    size_t off = 0;
    auto take = [&](size_t bytes) { char* p = ws + off; off = (off + bytes + 255) & ~(size_t)255; return p; };
    float*     hbuf   = (float*)take(NB);
    float*     xl     = (float*)take(NB);
    float*     xr     = (float*)take(NB);
    float*     xs     = (float*)take(NB);
    float*     agg    = (float*)take(NB);
    float*     ebuf   = (float*)take((size_t)NEDGES * sizeof(float));
    float*     denom  = (float*)take((size_t)NNODES * sizeof(float));
    unsigned*  mmax   = (unsigned*)take((size_t)NNODES * sizeof(unsigned));
    _Float16*  packed = (_Float16*)take((size_t)9 * 4 * 2 * 32 * 16 * sizeof(_Float16));

    gat_prepack<<<(9 * 4 * 2 * 32 * 16 + 255) / 256, 256, 0, stream>>>(Wl, Wr, Ws, packed);

    const float* hin = x;
    for (int layer = 0; layer < 3; ++layer) {
        gat_init<<<(NNODES * DIM + 255) / 256, 256, 0, stream>>>(agg, denom, mmax);
        gat_gemm3_wmma<<<(NTILES + 7) / 8, 256, 0, stream>>>(
            hin, packed + (size_t)layer * 3 * 4 * 2 * 32 * 16,
            bl + layer * DIM, br + layer * DIM, bs + layer * DIM, xl, xr, xs);
        gat_edge_attn<<<(NEDGES + 255) / 256, 256, 0, stream>>>(
            ei, xl, xr, att + layer * DIM, ebuf, mmax);
        gat_edge_exp<<<(NEDGES + 255) / 256, 256, 0, stream>>>(ei, mmax, ebuf, denom);
        gat_edge_agg<<<NEDGES / 8, 256, 0, stream>>>(ei, ebuf, denom, xl, agg);
        gat_node_post<<<(NNODES * 16 + 255) / 256, 256, 0, stream>>>(
            agg, xs, bg + layer * DIM, hbuf, layer < 2 ? 1 : 0);
        hin = hbuf;
    }
    gat_out_proj<<<(NNODES + 255) / 256, 256, 0, stream>>>(hbuf, Wout, bout, out);
}
